// MultiRNNCell_13091060318238
// MI455X (gfx1250) — compile-verified, matrix operand order checked by reading, not run on hardware
//
#include <hip/hip_runtime.h>
#include <math.h>

// GRU-D imputation RNN for MI455X (gfx1250), persistent-block design.
// B=64,T=1024,D=128,H=256. 4 blocks x 16 batch rows, bf16 WMMA, f32 accum.
// K-loops kept rolled (#pragma unroll 2) to avoid fragment spills to scratch.

#define B_  64
#define T_  1024
#define D_  128
#define H_  256

typedef __attribute__((ext_vector_type(16))) __bf16 v16bf;
typedef __attribute__((ext_vector_type(8)))  float  v8f;

static __device__ __forceinline__ unsigned short f2bf(float f) {
    union { float f; unsigned u; } x; x.f = f;
    unsigned r = x.u + 0x7FFFu + ((x.u >> 16) & 1u);   // round-to-nearest-even
    return (unsigned short)(r >> 16);
}
static __device__ __forceinline__ float bf2f(unsigned short h) {
    union { unsigned u; float f; } x; x.u = ((unsigned)h) << 16; return x.f;
}
static __device__ __forceinline__ float sigm_(float x) { return 1.f / (1.f + __expf(-x)); }

// A-fragment: 16x32 bf16 tile from LDS, row-major with `stride` halves.
// Lane L<16 holds row L, K-chunks {k0..k0+7} and {k0+16..k0+23};
// lane L>=16 holds row L-16, K-chunks {k0+8..} and {k0+24..} (ISA 7.12.2 layout).
static __device__ __forceinline__ v16bf lds_afrag(const unsigned short* base, int stride, int k0) {
    int lane = threadIdx.x & 31;
    const unsigned short* p = base + (lane & 15) * stride + k0 + (lane >> 4) * 8;
    union { uint4 q[2]; v16bf v; } u;
    u.q[0] = *(const uint4*)(p);
    u.q[1] = *(const uint4*)(p + 16);
    return u.v;
}
// B-fragment: 32x16 bf16 tile; weights stored row-major [N x K] (i.e. W, so
// B = W^T). Lane picks output column n, same two 16-byte K-chunks as A.
static __device__ __forceinline__ v16bf glb_bfrag(const unsigned short* w, int ldk, int k0) {
    int lane = threadIdx.x & 31;
    const unsigned short* p = w + (lane & 15) * ldk + k0 + (lane >> 4) * 8;
    union { uint4 q[2]; v16bf v; } u;
    u.q[0] = *(const uint4*)(p);
    u.q[1] = *(const uint4*)(p + 16);
    return u.v;
}
static __device__ __forceinline__ v8f wmma_bf16(v16bf a, v16bf b, v8f c) {
    return __builtin_amdgcn_wmma_f32_16x16x32_bf16(false, a, false, b, (short)0, c, false, false);
}

// ---------------- prep kernels (one-time per launch) ----------------

__global__ void k_f32_to_bf16(const float* __restrict__ s, unsigned short* __restrict__ d, int n) {
    int i = blockIdx.x * blockDim.x + threadIdx.x;
    if (i < n) d[i] = f2bf(s[i]);
}

// Layer1 combined weights [1024 x 256]:
// rows 0..511  : W_ih1+W_hh1 (r,z gates; input==hidden==h1)
// rows 512..767: W_ih1 n-rows (inn)   rows 768..1023: W_hh1 n-rows (hn)
__global__ void k_build_wc1(const float* __restrict__ Wih1, const float* __restrict__ Whh1,
                            unsigned short* __restrict__ dst) {
    int i = blockIdx.x * blockDim.x + threadIdx.x;
    if (i >= 1024 * 256) return;
    int n = i >> 8, k = i & 255;
    float v;
    if (n < 512)      v = Wih1[n * 256 + k] + Whh1[n * 256 + k];
    else if (n < 768) v = Wih1[n * 256 + k];
    else              v = Whh1[(n - 256) * 256 + k];
    dst[i] = f2bf(v);
}

// Pre-summed gate biases: [r_comb, z_comb, inn, hn] x 256 per layer, + b_out.
__global__ void k_build_bias(const float* bih0, const float* bhh0,
                             const float* bih1, const float* bhh1, const float* bo,
                             float* b0, float* b1, float* bow) {
    int i = blockIdx.x * blockDim.x + threadIdx.x;
    if (i < 1024) {
        int g = i >> 8, j = i & 255;
        float v;
        if (g == 0)      v = bih0[j] + bhh0[j];
        else if (g == 1) v = bih0[256 + j] + bhh0[256 + j];
        else if (g == 2) v = bih0[512 + j];
        else             v = bhh0[512 + j];
        b0[i] = v;
    } else if (i < 2048) {
        int q = i - 1024, g = q >> 8, j = q & 255;
        float v;
        if (g == 0)      v = bih1[j] + bhh1[j];
        else if (g == 1) v = bih1[256 + j] + bhh1[256 + j];
        else if (g == 2) v = bih1[512 + j];
        else             v = bhh1[512 + j];
        b1[q] = v;
    } else if (i < 2176) {
        bow[i - 2048] = bo[i - 2048];
    }
}

// ---------------- persistent recurrent kernel ----------------
// One block per 16 batch rows (one WMMA M-tile); 512 threads = 16 wave32s.

__global__ void __launch_bounds__(512) gru_persistent(
    const float* __restrict__ X, const float* __restrict__ Mm,
    const unsigned short* __restrict__ Wih0, const unsigned short* __restrict__ Whh0,
    const unsigned short* __restrict__ Wc1,  const unsigned short* __restrict__ Wout,
    const float* __restrict__ bias0, const float* __restrict__ bias1,
    const float* __restrict__ bout, float* __restrict__ out)
{
    __shared__ __align__(16) unsigned short sH [16 * 272];   // hidden (bf16), padded stride
    __shared__ __align__(16) unsigned short sH1[16 * 272];   // layer0 output
    __shared__ __align__(16) unsigned short sImp[16 * 144];  // imputed input (bf16)
    __shared__ float sEst[16 * 132];                         // est = previous step's out (f32)
    __shared__ float sB0[1024], sB1[1024], sBo[128];

    const int tid   = threadIdx.x;
    const int wid   = tid >> 5;
    const int lane  = tid & 31;
    const int mBase = blockIdx.x * 16;

    for (int i = tid; i < 16 * 272; i += 512) { sH[i] = 0; sH1[i] = 0; }
    for (int i = tid; i < 1024; i += 512) { sB0[i] = bias0[i]; sB1[i] = bias1[i]; }
    if (tid < 128) sBo[tid] = bout[tid];
    for (int i = tid; i < 16 * D_; i += 512) {               // h0 = 0 -> est = b_out
        int r = i >> 7, d = i & 127;
        sEst[r * 132 + d] = bout[d];
    }
    __syncthreads();

    const int col  = lane & 15;
    const int rAdd = (lane >> 4) * 8;                        // C/D frag: M = v + rAdd, N = col

    for (int t = 0; t < T_; ++t) {
        // ---- Phase A: imputed = m*x + (1-m)*est, bf16 into LDS ----
        for (int i = tid; i < 16 * D_; i += 512) {
            int r = i >> 7, d = i & 127;
            long g = ((long)(mBase + r) * T_ + t) * D_ + d;
            float xm = X[g], mm = Mm[g];
            float v = mm * xm + (1.f - mm) * sEst[r * 132 + d];
            sImp[r * 144 + d] = f2bf(v);
        }
        __syncthreads();

        // ---- Phase B: layer0 GRU; wave j owns H-column block j*16 ----
        {
            int j = wid;
            v8f aR = {}, aZ = {}, aN = {}, aHN = {};
            const unsigned short* wR = Wih0 + (0 * H_ + j * 16) * D_;
            const unsigned short* wZ = Wih0 + (1 * H_ + j * 16) * D_;
            const unsigned short* wN = Wih0 + (2 * H_ + j * 16) * D_;
            #pragma unroll 2
            for (int k0 = 0; k0 < D_; k0 += 32) {            // gi: A=imputed, K=128
                v16bf a = lds_afrag(sImp, 144, k0);
                aR = wmma_bf16(a, glb_bfrag(wR, D_, k0), aR);
                aZ = wmma_bf16(a, glb_bfrag(wZ, D_, k0), aZ);
                aN = wmma_bf16(a, glb_bfrag(wN, D_, k0), aN);
            }
            const unsigned short* uR = Whh0 + (0 * H_ + j * 16) * H_;
            const unsigned short* uZ = Whh0 + (1 * H_ + j * 16) * H_;
            const unsigned short* uN = Whh0 + (2 * H_ + j * 16) * H_;
            #pragma unroll 2
            for (int k0 = 0; k0 < H_; k0 += 32) {            // gh: A=h, K=256
                v16bf a = lds_afrag(sH, 272, k0);
                aR  = wmma_bf16(a, glb_bfrag(uR, H_, k0), aR);
                aZ  = wmma_bf16(a, glb_bfrag(uZ, H_, k0), aZ);
                aHN = wmma_bf16(a, glb_bfrag(uN, H_, k0), aHN);
            }
            int jc = j * 16 + col;
            #pragma unroll
            for (int v = 0; v < 8; ++v) {
                int m = v + rAdd;
                float r = sigm_(aR[v] + sB0[jc]);
                float z = sigm_(aZ[v] + sB0[256 + jc]);
                float n = tanhf(aN[v] + sB0[512 + jc] + r * (aHN[v] + sB0[768 + jc]));
                float hOld = bf2f(sH[m * 272 + jc]);
                sH1[m * 272 + jc] = f2bf((1.f - z) * n + z * hOld);
            }
        }
        __syncthreads();

        // ---- Phase C: layer1 GRU (input==hidden==h1, combined r/z weights) ----
        {
            int j = wid;
            v8f aR = {}, aZ = {}, aN = {}, aHN = {};
            const unsigned short* wR  = Wc1 + (0 * H_ + j * 16) * H_;
            const unsigned short* wZ  = Wc1 + (1 * H_ + j * 16) * H_;
            const unsigned short* wN  = Wc1 + (2 * H_ + j * 16) * H_;
            const unsigned short* wHN = Wc1 + (3 * H_ + j * 16) * H_;
            #pragma unroll 2
            for (int k0 = 0; k0 < H_; k0 += 32) {
                v16bf a = lds_afrag(sH1, 272, k0);
                aR  = wmma_bf16(a, glb_bfrag(wR,  H_, k0), aR);
                aZ  = wmma_bf16(a, glb_bfrag(wZ,  H_, k0), aZ);
                aN  = wmma_bf16(a, glb_bfrag(wN,  H_, k0), aN);
                aHN = wmma_bf16(a, glb_bfrag(wHN, H_, k0), aHN);
            }
            int jc = j * 16 + col;
            #pragma unroll
            for (int v = 0; v < 8; ++v) {
                int m = v + rAdd;
                float r = sigm_(aR[v] + sB1[jc]);
                float z = sigm_(aZ[v] + sB1[256 + jc]);
                float n = tanhf(aN[v] + sB1[512 + jc] + r * (aHN[v] + sB1[768 + jc]));
                float h1o = bf2f(sH1[m * 272 + jc]);
                sH[m * 272 + jc] = f2bf((1.f - z) * n + z * h1o);  // h <- h2
            }
        }
        __syncthreads();

        // ---- Phase D: out = h2 @ Wout^T + b_out; also becomes next est ----
        if (wid < 8) {
            int j = wid;
            v8f acc = {};
            const unsigned short* w = Wout + (j * 16) * H_;
            #pragma unroll 2
            for (int k0 = 0; k0 < H_; k0 += 32) {
                v16bf a = lds_afrag(sH, 272, k0);
                acc = wmma_bf16(a, glb_bfrag(w, H_, k0), acc);
            }
            int dc = j * 16 + col;
            #pragma unroll
            for (int v = 0; v < 8; ++v) {
                int m = v + rAdd;
                float val = acc[v] + sBo[dc];
                out[((long)(mBase + m) * T_ + t) * D_ + dc] = val;
                sEst[m * 132 + dc] = val;
            }
        } else if (t + 1 < T_) {
            // idle waves prefetch next timestep's X/mask (global_prefetch_b8)
            int r = (wid - 8) * 2 + (lane >> 4);
            long g = ((long)(mBase + r) * T_ + (t + 1)) * D_ + (lane & 15) * 8;
            __builtin_prefetch(&X[g], 0, 3);
            __builtin_prefetch(&Mm[g], 0, 3);
        }
        __syncthreads();
    }

    // final hidden state: d_out[B*T*D ..] = h_final [B x H]
    for (int i = tid; i < 16 * H_; i += 512) {
        int r = i >> 8, c = i & 255;
        out[(long)B_ * T_ * D_ + (mBase + r) * H_ + c] = bf2f(sH[r * 272 + c]);
    }
}

// ---------------- launch ----------------

extern "C" void kernel_launch(void* const* d_in, const int* in_sizes, int n_in,
                              void* d_out, int out_size, void* d_ws, size_t ws_size,
                              hipStream_t stream)
{
    const float* X    = (const float*)d_in[0];
    const float* Mm   = (const float*)d_in[1];
    const float* Wih0 = (const float*)d_in[2];
    const float* Whh0 = (const float*)d_in[3];
    const float* bih0 = (const float*)d_in[4];
    const float* bhh0 = (const float*)d_in[5];
    const float* Wih1 = (const float*)d_in[6];
    const float* Whh1 = (const float*)d_in[7];
    const float* bih1 = (const float*)d_in[8];
    const float* bhh1 = (const float*)d_in[9];
    const float* Wout = (const float*)d_in[10];
    const float* bout = (const float*)d_in[11];

    char* ws = (char*)d_ws;                       // ~1.19 MB used
    unsigned short* wih0b = (unsigned short*)(ws + 0);        // 768x128 bf16
    unsigned short* whh0b = (unsigned short*)(ws + 196608);   // 768x256 bf16
    unsigned short* wc1b  = (unsigned short*)(ws + 589824);   // 1024x256 bf16
    unsigned short* woutb = (unsigned short*)(ws + 1114112);  // 128x256 bf16
    float* bias0 = (float*)(ws + 1179648);
    float* bias1 = (float*)(ws + 1183744);
    float* boutw = (float*)(ws + 1187840);

    k_f32_to_bf16<<<(768 * 128 + 255) / 256, 256, 0, stream>>>(Wih0, wih0b, 768 * 128);
    k_f32_to_bf16<<<(768 * 256 + 255) / 256, 256, 0, stream>>>(Whh0, whh0b, 768 * 256);
    k_f32_to_bf16<<<(128 * 256 + 255) / 256, 256, 0, stream>>>(Wout, woutb, 128 * 256);
    k_build_wc1<<<(1024 * 256 + 255) / 256, 256, 0, stream>>>(Wih1, Whh1, wc1b);
    k_build_bias<<<(2176 + 255) / 256, 256, 0, stream>>>(bih0, bhh0, bih1, bhh1, bout,
                                                         bias0, bias1, boutw);

    gru_persistent<<<4, 512, 0, stream>>>(X, Mm, wih0b, whh0b, wc1b, woutb,
                                          bias0, bias1, boutw, (float*)d_out);
}
